// _Block_37288906064244
// MI455X (gfx1250) — matrix operand order repaired and checked by reading, compiled, not run on hardware
//
#include <hip/hip_runtime.h>
#include <hip/hip_bf16.h>

#define DIM   1024
#define HEADS 16
#define HD    64
#define SEQ   2048
#define BATCH 2
#define NTOK  (BATCH * SEQ)   // 4096
#define MLP   4096

typedef __bf16 bf16_t;
typedef __attribute__((ext_vector_type(16))) __bf16 v16bf;
typedef __attribute__((ext_vector_type(8)))  __bf16 v8bf;
typedef __attribute__((ext_vector_type(4)))  __bf16 v4bf;
typedef __attribute__((ext_vector_type(8)))  float  v8f;
typedef unsigned int u32;
typedef __attribute__((ext_vector_type(4))) u32 u32x4;
typedef __attribute__((ext_vector_type(4))) int i32x4;
typedef __attribute__((ext_vector_type(8))) int i32x8;

static __device__ inline int lane_id() { return threadIdx.x & 31; }

static __device__ inline v8f vzero8() {
  v8f z = {0.f, 0.f, 0.f, 0.f, 0.f, 0.f, 0.f, 0.f};
  return z;
}

static __device__ inline v8f wmma_bf16(v16bf a, v16bf b, v8f c) {
  return __builtin_amdgcn_wmma_f32_16x16x32_bf16(
      false, a, false, b, (short)0, c, false, false);
}

// ---------------- Tensor Data Mover: 2-D tile load, bf16 elements ----------------
// Loads a (rows x 32-element) bf16 tile with source row pitch `stride_elems`
// into LDS at byte offset `lds_off`; TDM pad feature inserts 16B after every
// 64B row (pad_interval=16 dw, pad_amount=4 dw) -> LDS row pitch 80B (40 bf16).
static __device__ inline void tdm_load_tile32(u32 lds_off, const bf16_t* gsrc,
                                              u32 stride_elems, u32 rows,
                                              u32 tensor_rows) {
  unsigned long long ga = (unsigned long long)(uintptr_t)gsrc;
  u32x4 g0;
  g0[0] = 1u;                                   // count=1, user descriptor
  g0[1] = lds_off;                              // lds_addr (bytes)
  g0[2] = (u32)(ga & 0xFFFFFFFFu);              // global_addr[31:0]
  g0[3] = (u32)((ga >> 32) & 0x01FFFFFFu) | (2u << 30);  // addr[56:32] | type=2
  i32x8 g1;
  g1[0] = (1 << 16) | (1 << 20) | (3 << 22) | (3 << 25);
          // data_size=1 (2B) | pad_enable | pad_interval=16dw | pad_amount=4dw
  g1[1] = (int)((stride_elems & 0xFFFFu) << 16);          // tensor_dim0[15:0]
  g1[2] = (int)(((stride_elems >> 16) & 0xFFFFu) |
                ((tensor_rows & 0xFFFFu) << 16));         // dim0[31:16]|dim1[15:0]
  g1[3] = (int)(((tensor_rows >> 16) & 0xFFFFu) |
                (32u << 16));                             // dim1[31:16]|tile_dim0=32
  g1[4] = (int)(rows & 0xFFFFu);                          // tile_dim1 (tile_dim2=0)
  g1[5] = (int)stride_elems;                              // tensor_dim0_stride[31:0]
  g1[6] = 0;
  g1[7] = 0;
  i32x4 z4 = {0, 0, 0, 0};
#if defined(__clang_major__) && (__clang_major__ >= 23)
  i32x8 z8 = {0, 0, 0, 0, 0, 0, 0, 0};
  __builtin_amdgcn_tensor_load_to_lds(g0, g1, z4, z4, z8, 0);
#else
  __builtin_amdgcn_tensor_load_to_lds(g0, g1, z4, z4, 0);
#endif
}

static __device__ inline u32 lds_byte_off(const void* p) {
  return (u32)(uintptr_t)p;   // low 32 bits of a flat LDS address = LDS offset
}

// ---------------- fragment loaders (bf16 sources) ----------------
static __device__ inline v16bf frag_a_lds(const bf16_t* base, int ld, int k0) {
  const int l = lane_id();
  const int off = (l & 16) ? 8 : 0;
  const bf16_t* p = base + (l & 15) * ld + k0 + off;
  const v8bf* p8 = (const v8bf*)p;
  v8bf lo = p8[0];
  v8bf hi = p8[2];
  v16bf f;
#pragma unroll
  for (int i = 0; i < 8; ++i) { f[i] = lo[i]; f[8 + i] = hi[i]; }
  return f;
}

static __device__ inline v16bf frag_b_lds(const bf16_t* base, int ld, int col0, int k0) {
  const int l = lane_id();
  const int off = (l & 16) ? 8 : 0;
  const bf16_t* p = base + (col0 + (l & 15)) * ld + k0 + off;
  const v8bf* p8 = (const v8bf*)p;
  v8bf lo = p8[0], hi = p8[2];
  v16bf f;
#pragma unroll
  for (int i = 0; i < 8; ++i) { f[i] = lo[i]; f[8 + i] = hi[i]; }
  return f;
}

static __device__ inline v16bf frag_a_g(const bf16_t* __restrict__ base, int ld, int k0) {
  const int l = lane_id();
  const int off = (l & 16) ? 8 : 0;
  const bf16_t* p = base + (size_t)(l & 15) * ld + k0 + off;
  v8bf lo = *(const v8bf*)p;
  v8bf hi = *(const v8bf*)(p + 16);
  v16bf f;
#pragma unroll
  for (int i = 0; i < 8; ++i) { f[i] = lo[i]; f[8 + i] = hi[i]; }
  return f;
}

// ---------------- fp32 -> bf16 conversion (weights) ----------------
__global__ __launch_bounds__(256) void cvt_kernel(const float* __restrict__ s,
                                                  bf16_t* __restrict__ d, int n) {
  int i = (blockIdx.x * 256 + threadIdx.x) * 4;
  if (i < n) {
    float4 v = *(const float4*)(s + i);
    v4bf o = {(bf16_t)v.x, (bf16_t)v.y, (bf16_t)v.z, (bf16_t)v.w};
    *(v4bf*)(d + i) = o;
  }
}

// ---------------- LayerNorm (fp32 in, bf16 out) ----------------
__global__ __launch_bounds__(256) void ln_kernel(const float* __restrict__ x,
                                                 const float* __restrict__ w,
                                                 const float* __restrict__ b,
                                                 bf16_t* __restrict__ y) {
  const float* xr = x + (size_t)blockIdx.x * DIM;
  bf16_t* yr = y + (size_t)blockIdx.x * DIM;
  float4 v = ((const float4*)xr)[threadIdx.x];
  float s  = v.x + v.y + v.z + v.w;
  float sq = v.x * v.x + v.y * v.y + v.z * v.z + v.w * v.w;
  __shared__ float rs[256], rq[256];
  rs[threadIdx.x] = s; rq[threadIdx.x] = sq;
  __syncthreads();
#pragma unroll
  for (int stride = 128; stride > 0; stride >>= 1) {
    if (threadIdx.x < stride) {
      rs[threadIdx.x] += rs[threadIdx.x + stride];
      rq[threadIdx.x] += rq[threadIdx.x + stride];
    }
    __syncthreads();
  }
  const float mu  = rs[0] * (1.0f / DIM);
  const float var = rq[0] * (1.0f / DIM) - mu * mu;
  const float rsd = rsqrtf(var + 1e-5f);
  float4 wv = ((const float4*)w)[threadIdx.x];
  float4 bv = ((const float4*)b)[threadIdx.x];
  v4bf o = {(bf16_t)((v.x - mu) * rsd * wv.x + bv.x),
            (bf16_t)((v.y - mu) * rsd * wv.y + bv.y),
            (bf16_t)((v.z - mu) * rsd * wv.z + bv.z),
            (bf16_t)((v.w - mu) * rsd * wv.w + bv.w)};
  ((v4bf*)yr)[threadIdx.x] = o;
}

// ---------------- GEMM: C = act(A[M,K] * W[N,K]^T + bias) (+ Res) ----------------
// 128 threads = 4 waves (2 in M x 2 in N); block tile 128x128, wave tile 64x64
// (4x4 WMMA accumulators -> 16 WMMAs per 8 LDS fragment loads). A/W tiles
// staged by the Tensor Data Mover into double-buffered LDS (bf16), one TDM
// descriptor per wave per k-step (waves 0-1: 64 A rows, waves 2-3: 64 W rows).
template <int ACT, bool RES, typename OutT>
__global__ __launch_bounds__(128) void gemm_kernel(const bf16_t* __restrict__ A,
                                                   const bf16_t* __restrict__ W,
                                                   const float* __restrict__ bias,
                                                   const float* __restrict__ Res,
                                                   OutT* __restrict__ C,
                                                   int M, int N, int K) {
  __shared__ __align__(16) bf16_t As[2][128][40];
  __shared__ __align__(16) bf16_t Ws[2][128][40];
  const int wid = threadIdx.x >> 5;
  const int waveM = wid & 1, waveN = wid >> 1;
  const int mblk = blockIdx.y * 128, nblk = blockIdx.x * 128;

  // per-wave TDM slice: waves 0-1 stage 64 A rows, waves 2-3 stage 64 W rows
  const bool isA = (wid < 2);
  const int slice = isA ? (64 * wid) : (64 * (wid - 2));
  const bf16_t* src = isA ? (A + (size_t)(mblk + slice) * K)
                          : (W + (size_t)(nblk + slice) * K);
  const u32 dst_off[2] = {
      lds_byte_off(isA ? &As[0][slice][0] : &Ws[0][slice][0]),
      lds_byte_off(isA ? &As[1][slice][0] : &Ws[1][slice][0])};
  const u32 trows = (u32)(isA ? M : N) - (u32)(isA ? mblk : nblk) - (u32)slice;

  tdm_load_tile32(dst_off[0], src, (u32)K, 64u, trows);
  __builtin_amdgcn_s_wait_tensorcnt(0);
  __syncthreads();

  v8f acc[4][4];
#pragma unroll
  for (int i = 0; i < 4; ++i)
#pragma unroll
    for (int j = 0; j < 4; ++j) acc[i][j] = vzero8();

  int cur = 0;
  for (int k0 = 0; k0 < K; k0 += 32) {
    if (k0 + 32 < K)
      tdm_load_tile32(dst_off[cur ^ 1], src + (k0 + 32), (u32)K, 64u, trows);

    v16bf af[4], bfr[4];
#pragma unroll
    for (int i = 0; i < 4; ++i)
      af[i] = frag_a_lds(&As[cur][waveM * 64 + 16 * i][0], 40, 0);
#pragma unroll
    for (int j = 0; j < 4; ++j)
      bfr[j] = frag_b_lds(&Ws[cur][0][0], 40, waveN * 64 + 16 * j, 0);
#pragma unroll
    for (int i = 0; i < 4; ++i)
#pragma unroll
      for (int j = 0; j < 4; ++j)
        acc[i][j] = wmma_bf16(af[i], bfr[j], acc[i][j]);

    if (k0 + 32 < K) __builtin_amdgcn_s_wait_tensorcnt(0);
    __syncthreads();
    cur ^= 1;
  }

  const int l = lane_id();
  const int mo = (l & 16) ? 8 : 0;
  const int nc = l & 15;
  const int m0 = mblk + waveM * 64;
  const int n0 = nblk + waveN * 64;
#pragma unroll
  for (int i = 0; i < 4; ++i) {
#pragma unroll
    for (int j = 0; j < 4; ++j) {
      const int n = n0 + 16 * j + nc;
      const float bn = bias[n];
#pragma unroll
      for (int r = 0; r < 8; ++r) {
        const int m = m0 + 16 * i + r + mo;
        float v = acc[i][j][r] + bn;
        if (ACT == 1) v = 0.5f * v * (1.0f + erff(v * 0.70710678118654752f));
        if (RES) v += Res[(size_t)m * N + n];
        C[(size_t)m * N + n] = (OutT)v;
      }
    }
  }
}

// ---------------- Flash attention (bf16 qkv in, bf16 O out) ----------------
// grid = (SEQ/128, HEADS, BATCH); block = 256 = 8 waves, 16 queries/wave.
__global__ __launch_bounds__(256) void attn_kernel(const bf16_t* __restrict__ qkv,
                                                   bf16_t* __restrict__ O) {
  const int bz = blockIdx.z;
  const int h  = blockIdx.y;
  const int wid = threadIdx.x >> 5;
  const int l = lane_id();
  const int q0 = blockIdx.x * 128 + wid * 16;

  __shared__ __align__(16) bf16_t Ks[32][72];      // [key][hd]
  __shared__ __align__(16) bf16_t Vt[64][40];      // [hd][key] transposed
  __shared__ __align__(16) bf16_t Ps[8][16][40];   // per-wave P tile [q][key]

  const bf16_t* qbase = qkv + ((size_t)(bz * SEQ + q0)) * (3 * DIM) + h * HD;
  const v16bf qa0 = frag_a_g(qbase, 3 * DIM, 0);
  const v16bf qa1 = frag_a_g(qbase, 3 * DIM, 32);

  v8f oacc[4];
#pragma unroll
  for (int t = 0; t < 4; ++t) oacc[t] = vzero8();
  float mrun[8], lrun[8];
#pragma unroll
  for (int r = 0; r < 8; ++r) { mrun[r] = -3.0e38f; lrun[r] = 0.f; }

  const int kr = threadIdx.x >> 3;        // 0..31: key row staged by this thread
  const int d0 = (threadIdx.x & 7) * 8;   // 8 head-dims staged by this thread
  const int mo = (l & 16) ? 8 : 0;
  bf16_t* pbase = &Ps[wid][0][0];

  for (int kt = 0; kt < SEQ; kt += 32) {
    { // stage K and V (V transposed) cooperatively
      const bf16_t* kp = qkv + ((size_t)(bz * SEQ + kt + kr)) * (3 * DIM) + DIM + h * HD + d0;
      v8bf kk = *(const v8bf*)kp;
      v8bf vv = *(const v8bf*)(kp + DIM);
#pragma unroll
      for (int i = 0; i < 8; ++i) {
        Ks[kr][d0 + i] = kk[i];
        Vt[d0 + i][kr] = vv[i];
      }
    }
    __syncthreads();

    // S = Q * K^T (two 16-key tiles, contraction over hd=64)
    v8f s0 = vzero8(), s1 = vzero8();
    s0 = wmma_bf16(qa0, frag_b_lds(&Ks[0][0], 72, 0, 0), s0);
    s0 = wmma_bf16(qa1, frag_b_lds(&Ks[0][0], 72, 0, 32), s0);
    s1 = wmma_bf16(qa0, frag_b_lds(&Ks[0][0], 72, 16, 0), s1);
    s1 = wmma_bf16(qa1, frag_b_lds(&Ks[0][0], 72, 16, 32), s1);

    // online softmax in D-register layout (row m = r + mo, key col = l&15)
#pragma unroll
    for (int r = 0; r < 8; ++r) {
      float v0 = s0[r] * 0.125f, v1 = s1[r] * 0.125f;   // scale = hd^-0.5
      float mx = fmaxf(v0, v1);
#pragma unroll
      for (int sh = 8; sh >= 1; sh >>= 1) mx = fmaxf(mx, __shfl_xor(mx, sh, 32));
      const float mnew = fmaxf(mrun[r], mx);
      const float corr = __expf(mrun[r] - mnew);
      const float p0 = __expf(v0 - mnew);
      const float p1 = __expf(v1 - mnew);
      float rsum = p0 + p1;
#pragma unroll
      for (int sh = 8; sh >= 1; sh >>= 1) rsum += __shfl_xor(rsum, sh, 32);
      lrun[r] = lrun[r] * corr + rsum;
      mrun[r] = mnew;
#pragma unroll
      for (int t = 0; t < 4; ++t) oacc[t][r] *= corr;
      const int m = r + mo;
      pbase[m * 40 + (l & 15)]      = (bf16_t)p0;
      pbase[m * 40 + 16 + (l & 15)] = (bf16_t)p1;
    }

    // O += P * V (wave-private P restaged through LDS into A-fragment layout)
    const v16bf pa = frag_a_lds(pbase, 40, 0);
#pragma unroll
    for (int t = 0; t < 4; ++t) {
      v16bf vb = frag_b_lds(&Vt[0][0], 40, 16 * t, 0);
      oacc[t] = wmma_bf16(pa, vb, oacc[t]);
    }
    __syncthreads();
  }

#pragma unroll
  for (int r = 0; r < 8; ++r) {
    const float inv = 1.0f / lrun[r];
    const int q = q0 + r + mo;
    bf16_t* op = O + ((size_t)(bz * SEQ + q)) * DIM + h * HD;
#pragma unroll
    for (int t = 0; t < 4; ++t) op[16 * t + (l & 15)] = (bf16_t)(oacc[t][r] * inv);
  }
}

// ---------------- driver ----------------
extern "C" void kernel_launch(void* const* d_in, const int* in_sizes, int n_in,
                              void* d_out, int out_size, void* d_ws, size_t ws_size,
                              hipStream_t stream) {
  const float* x      = (const float*)d_in[0];
  const float* ln1_w  = (const float*)d_in[1];
  const float* ln1_b  = (const float*)d_in[2];
  const float* ln2_w  = (const float*)d_in[3];
  const float* ln2_b  = (const float*)d_in[4];
  const float* qkv_w  = (const float*)d_in[5];
  const float* qkv_b  = (const float*)d_in[6];
  const float* proj_w = (const float*)d_in[7];
  const float* proj_b = (const float*)d_in[8];
  const float* fc1_w  = (const float*)d_in[9];
  const float* fc1_b  = (const float*)d_in[10];
  const float* fc2_w  = (const float*)d_in[11];
  const float* fc2_b  = (const float*)d_in[12];
  float* out = (float*)d_out;

  char* p = (char*)d_ws;
  bf16_t* xn    = (bf16_t*)p;  p += (size_t)NTOK * DIM * 2;          // [4096,1024] bf16
  bf16_t* qkvb  = (bf16_t*)p;  p += (size_t)NTOK * 3 * DIM * 2;      // [4096,3072] bf16
  bf16_t* ob    = (bf16_t*)p;  p += (size_t)NTOK * DIM * 2;          // [4096,1024] bf16
  float*  x1    = (float*)p;   p += (size_t)NTOK * DIM * 4;          // [4096,1024] f32
  bf16_t* hb    = (bf16_t*)p;  p += (size_t)NTOK * MLP * 2;          // [4096,4096] bf16
  bf16_t* wqkv  = (bf16_t*)p;  p += (size_t)3 * DIM * DIM * 2;
  bf16_t* wproj = (bf16_t*)p;  p += (size_t)DIM * DIM * 2;
  bf16_t* wfc1  = (bf16_t*)p;  p += (size_t)MLP * DIM * 2;
  bf16_t* wfc2  = (bf16_t*)p;  p += (size_t)DIM * MLP * 2;

  // 0) per-launch weight downconversion to bf16
  {
    int n;
    n = 3 * DIM * DIM; cvt_kernel<<<(n / 4 + 255) / 256, 256, 0, stream>>>(qkv_w, wqkv, n);
    n = DIM * DIM;     cvt_kernel<<<(n / 4 + 255) / 256, 256, 0, stream>>>(proj_w, wproj, n);
    n = MLP * DIM;     cvt_kernel<<<(n / 4 + 255) / 256, 256, 0, stream>>>(fc1_w, wfc1, n);
    n = DIM * MLP;     cvt_kernel<<<(n / 4 + 255) / 256, 256, 0, stream>>>(fc2_w, wfc2, n);
  }

  // 1) LN1 -> bf16
  ln_kernel<<<NTOK, 256, 0, stream>>>(x, ln1_w, ln1_b, xn);
  // 2) QKV GEMM (bf16 out)
  gemm_kernel<0, false, bf16_t><<<dim3(3 * DIM / 128, NTOK / 128), 128, 0, stream>>>(
      xn, wqkv, qkv_b, nullptr, qkvb, NTOK, 3 * DIM, DIM);
  // 3) attention (bf16 out)
  attn_kernel<<<dim3(SEQ / 128, HEADS, BATCH), 256, 0, stream>>>(qkvb, ob);
  // 4) proj GEMM + residual (f32 out)
  gemm_kernel<0, true, float><<<dim3(DIM / 128, NTOK / 128), 128, 0, stream>>>(
      ob, wproj, proj_b, x, x1, NTOK, DIM, DIM);
  // 5) LN2 -> bf16
  ln_kernel<<<NTOK, 256, 0, stream>>>(x1, ln2_w, ln2_b, xn);
  // 6) FC1 GEMM + GELU (bf16 out)
  gemm_kernel<1, false, bf16_t><<<dim3(MLP / 128, NTOK / 128), 128, 0, stream>>>(
      xn, wfc1, fc1_b, nullptr, hb, NTOK, MLP, DIM);
  // 7) FC2 GEMM + residual -> out (f32)
  gemm_kernel<0, true, float><<<dim3(DIM / 128, NTOK / 128), 128, 0, stream>>>(
      hb, wfc2, fc2_b, x1, out, NTOK, DIM, MLP);
}
